// AFP_jittable_55731495632974
// MI455X (gfx1250) — matrix operand order repaired and checked by reading, not compiled, not run on hardware
//
#include <hip/hip_runtime.h>
#include <hip/hip_bf16.h>

// ---------------------------------------------------------------------------
// Types for CDNA5 WMMA (wave32): D(16x16,f32) = A(16x32,bf16) * B(32x16,bf16) + C
// ---------------------------------------------------------------------------
typedef __attribute__((ext_vector_type(16))) __bf16 v16bf;
typedef __attribute__((ext_vector_type(8)))  float  v8f;
typedef unsigned short u16;
typedef __attribute__((ext_vector_type(8)))  u16 v8u;
typedef __attribute__((ext_vector_type(16))) u16 v16u;

#define KPAD 160   // max K staged per LDS block (multiple of 32)

__device__ __forceinline__ u16 f2bf(float f) {
    unsigned u = __float_as_uint(f);
    unsigned r = (u + 0x7FFFu + ((u >> 16) & 1u)) >> 16;   // round-to-nearest-even
    return (u16)r;
}

// A fragment: lane holds row (lane&15); halves 0-7 -> K = 8*(lane>>4)+0..7,
// halves 8-15 -> K = 16 + 8*(lane>>4)+0..7  (two 16B contiguous LDS loads)
__device__ __forceinline__ v16bf load_a16(const u16* p) {
    v8u lo = *(const v8u*)p;
    v8u hi = *(const v8u*)(p + 16);
    v16u r;
#pragma unroll
    for (int i = 0; i < 8; ++i) { r[i] = lo[i]; r[8 + i] = hi[i]; }
    return __builtin_bit_cast(v16bf, r);
}
// B fragment: lane holds col (lane&15); halves i -> K = 16*(lane>>4)+i (one 32B load)
__device__ __forceinline__ v16bf load_b16(const u16* p) {
    v16u r = *(const v16u*)p;
    return __builtin_bit_cast(v16bf, r);
}

__device__ __forceinline__ float leaky01(float x) { return x >= 0.f ? x : 0.01f * x; }

// ---------------------------------------------------------------------------
// Generic WMMA GEMM: Y[M,N] = act( X[M,K] * W[N,K]^T + bias[N] )
// 256 threads = 8 waves; each wave owns 16 rows; N in 64-col LDS blocks,
// K in 160-element LDS blocks.  A fragments are hoisted into registers per
// K-block and reused across all 4 N-tiles.  ACT: 0=none 1=leaky(0.01) 2=relu.
// INRELU applies relu to X while staging (for the MLP head chain).
// ---------------------------------------------------------------------------
template <int ACT, bool INRELU>
__global__ void k_gemm(const float* __restrict__ X, const float* __restrict__ W,
                       const float* __restrict__ bias, float* __restrict__ Y,
                       int M, int N, int K) {
    __shared__ __align__(64) u16 Xs[128 * KPAD];
    __shared__ __align__(64) u16 Ws[64 * KPAD];
    const int tid  = threadIdx.x;
    const int r0   = blockIdx.x * 128;
    const int wave = tid >> 5, lane = tid & 31;
    const int lg   = lane >> 4, ln = lane & 15;

    for (int nb = 0; nb < N; nb += 64) {
        v8f acc[4];
#pragma unroll
        for (int q = 0; q < 4; ++q) { v8f z = {}; acc[q] = z; }

        for (int kb = 0; kb < K; kb += KPAD) {
            __syncthreads();
            if (nb == 0 || K > KPAD) {
                for (int idx = tid; idx < 128 * KPAD; idx += 256) {
                    int r = idx / KPAD, k = idx - r * KPAD;
                    int row = r0 + r, kk = kb + k;
                    float v = (row < M && kk < K) ? X[(size_t)row * K + kk] : 0.f;
                    if (INRELU) v = v > 0.f ? v : 0.f;
                    Xs[idx] = f2bf(v);
                }
            }
            for (int idx = tid; idx < 64 * KPAD; idx += 256) {
                int n = idx / KPAD, k = idx - n * KPAD;
                int col = nb + n, kk = kb + k;
                float v = (col < N && kk < K) ? W[(size_t)col * K + kk] : 0.f;
                Ws[idx] = f2bf(v);
            }
            __syncthreads();
            const int nc = min(5, (K - kb + 31) >> 5);
            // hoist A fragments: fixed rows per wave, reused for all 4 N-tiles
            v16bf afrag[5];
#pragma unroll
            for (int c = 0; c < 5; ++c) {
                if (c >= nc) break;
                afrag[c] = load_a16(&Xs[(wave * 16 + ln) * KPAD + c * 32 + 8 * lg]);
            }
#pragma unroll
            for (int nt = 0; nt < 4; ++nt) {
                if (nb + nt * 16 >= N) break;
                v8f a_ = acc[nt];
#pragma unroll
                for (int c = 0; c < 5; ++c) {
                    if (c >= nc) break;
                    v16bf b = load_a16(&Ws[(nt * 16 + ln) * KPAD + c * 32 + 8 * lg]);
                    a_ = __builtin_amdgcn_wmma_f32_16x16x32_bf16(
                        false, afrag[c], false, b, (short)0, a_, false, false);
                }
                acc[nt] = a_;
            }
        }
#pragma unroll
        for (int nt = 0; nt < 4; ++nt) {
            int col = nb + nt * 16 + ln;
            if (nb + nt * 16 >= N || col >= N) continue;
            float bcol = bias ? bias[col] : 0.f;
#pragma unroll
            for (int j = 0; j < 8; ++j) {
                int row = r0 + wave * 16 + j + 8 * lg;
                if (row < M) {
                    float y = acc[nt][j] + bcol;
                    if (ACT == 1) y = leaky01(y);
                    else if (ACT == 2) y = y > 0.f ? y : 0.f;
                    Y[(size_t)row * N + col] = y;
                }
            }
        }
    }
}

// ---------------------------------------------------------------------------
// Fused GATEConv edge-attention logits:
//   z[e,:] = concat(h0[src[e]], ea[e]) @ W1^T  (via WMMA, 16 edges/wave)
//   elog[e] = leaky( sum_n leaky(z[e,n]) * attl[n] + hdotr[dst[e]] )
// Never materializes the [400K,128] xj matrix.  A fragments loaded once and
// reused across all 8 N-tiles.
// ---------------------------------------------------------------------------
__global__ void k_edge_gate_logits(const float* __restrict__ h0,
                                   const float* __restrict__ ea,
                                   const int* __restrict__ src,
                                   const int* __restrict__ dst,
                                   const float* __restrict__ W1,   // [128,140]
                                   const float* __restrict__ attl, // [128]
                                   const float* __restrict__ hdotr,
                                   float* __restrict__ elog, int nE) {
    __shared__ __align__(64) u16 Xs[128 * KPAD];
    __shared__ __align__(64) u16 Ws[64 * KPAD];
    const int tid  = threadIdx.x;
    const int e0   = blockIdx.x * 128;
    const int wave = tid >> 5, lane = tid & 31;
    const int lg   = lane >> 4, ln = lane & 15;

    for (int idx = tid; idx < 128 * KPAD; idx += 256) {
        int r = idx / KPAD, k = idx - r * KPAD;
        int e = e0 + r;
        float v = 0.f;
        if (e < nE) {
            if (k < 128)      v = h0[(size_t)src[e] * 128 + k];
            else if (k < 140) v = ea[(size_t)e * 12 + (k - 128)];
        }
        Xs[idx] = f2bf(v);
    }
    __syncthreads();
    v16bf afrag[5];
#pragma unroll
    for (int c = 0; c < 5; ++c)
        afrag[c] = load_a16(&Xs[(wave * 16 + ln) * KPAD + c * 32 + 8 * lg]);

    float p[8];
#pragma unroll
    for (int j = 0; j < 8; ++j) p[j] = 0.f;

    for (int nb = 0; nb < 128; nb += 64) {
        __syncthreads();   // previous N-block's B reads complete before restage
        for (int idx = tid; idx < 64 * KPAD; idx += 256) {
            int n = idx / KPAD, k = idx - n * KPAD;
            float v = (k < 140) ? W1[(size_t)(nb + n) * 140 + k] : 0.f;
            Ws[idx] = f2bf(v);
        }
        __syncthreads();
#pragma unroll
        for (int nt = 0; nt < 4; ++nt) {
            v8f acc = {};
#pragma unroll
            for (int c = 0; c < 5; ++c) {
                v16bf b = load_a16(&Ws[(nt * 16 + ln) * KPAD + c * 32 + 8 * lg]);
                acc = __builtin_amdgcn_wmma_f32_16x16x32_bf16(
                    false, afrag[c], false, b, (short)0, acc, false, false);
            }
            float al = attl[nb + nt * 16 + ln];
#pragma unroll
            for (int j = 0; j < 8; ++j) p[j] += leaky01(acc[j]) * al;
        }
    }
    // reduce the 16 column-partials per row (rows split: lanes 0-15 hold M=j, 16-31 hold M=8+j)
#pragma unroll
    for (int off = 8; off >= 1; off >>= 1)
#pragma unroll
        for (int j = 0; j < 8; ++j) p[j] += __shfl_xor(p[j], off, 32);
    if (ln == 0) {
#pragma unroll
        for (int j = 0; j < 8; ++j) {
            int e = e0 + wave * 16 + j + 8 * lg;
            if (e < nE) elog[e] = leaky01(p[j] + hdotr[dst[e]]);
        }
    }
}

// ---------------------------------------------------------------------------
// Elementwise / segment helpers
// ---------------------------------------------------------------------------
__global__ void k_fill_f32(float* p, float v, int n) {
    int i = blockIdx.x * blockDim.x + threadIdx.x; if (i < n) p[i] = v;
}
__global__ void k_fill_u32(unsigned* p, unsigned v, int n) {
    int i = blockIdx.x * blockDim.x + threadIdx.x; if (i < n) p[i] = v;
}
__global__ void k_bias_init(float* m, const float* __restrict__ bias, int rows) {
    int i = blockIdx.x * blockDim.x + threadIdx.x;
    if (i < rows * 128) m[i] = bias[i & 127];
}
__global__ void k_elu(float* p, int n) {
    int i = blockIdx.x * blockDim.x + threadIdx.x;
    if (i < n) { float x = p[i]; p[i] = x > 0.f ? x : __expf(x) - 1.f; }
}
__global__ void k_relu(float* p, int n) {
    int i = blockIdx.x * blockDim.x + threadIdx.x;
    if (i < n) { float x = p[i]; p[i] = x > 0.f ? x : 0.f; }
}

// warp-per-row dual dot: o1 = X.v1, o2 = X.v2 (v2/o2 optional)
__global__ void k_node_dot2(const float* __restrict__ X, const float* __restrict__ v1,
                            const float* __restrict__ v2, float* __restrict__ o1,
                            float* __restrict__ o2, int rows) {
    int g = blockIdx.x * blockDim.x + threadIdx.x;
    int row = g >> 5, lane = g & 31;
    if (row >= rows) return;
    float s1 = 0.f, s2 = 0.f;
    for (int c = lane; c < 128; c += 32) {
        float x = X[(size_t)row * 128 + c];
        s1 += x * v1[c];
        if (v2) s2 += x * v2[c];
    }
#pragma unroll
    for (int off = 16; off >= 1; off >>= 1) {
        s1 += __shfl_xor(s1, off, 32);
        if (v2) s2 += __shfl_xor(s2, off, 32);
    }
    if (lane == 0) { o1[row] = s1; if (v2) o2[row] = s2; }
}

// elog[i] = leaky( asrc[srcIdx?srcIdx[i]:i] + adst[dstIdx[i]] )
__global__ void k_att_logits(const float* __restrict__ asrc, const float* __restrict__ adst,
                             const int* __restrict__ srcIdx, const int* __restrict__ dstIdx,
                             float* __restrict__ elog, int n) {
    int i = blockIdx.x * blockDim.x + threadIdx.x;
    if (i >= n) return;
    float a = asrc[srcIdx ? srcIdx[i] : i] + adst[dstIdx[i]];
    elog[i] = leaky01(a);
}

// monotonic uint encoding of float for atomic max
__device__ __forceinline__ unsigned fenc(float f) {
    unsigned u = __float_as_uint(f);
    return (u >> 31) ? ~u : (u | 0x80000000u);
}
__device__ __forceinline__ float fdec(unsigned u) {
    return (u >> 31) ? __uint_as_float(u & 0x7FFFFFFFu) : __uint_as_float(~u);
}
__global__ void k_seg_max(const float* __restrict__ elog, const int* __restrict__ dst,
                          unsigned* __restrict__ nmax, int n) {
    int i = blockIdx.x * blockDim.x + threadIdx.x;
    if (i < n) atomicMax(&nmax[dst[i]], fenc(elog[i]));
}
__global__ void k_seg_expsum(const float* __restrict__ elog, const int* __restrict__ dst,
                             const unsigned* __restrict__ nmax, float* __restrict__ eexp,
                             float* __restrict__ nden, int n) {
    int i = blockIdx.x * blockDim.x + threadIdx.x;
    if (i >= n) return;
    float m  = fdec(nmax[dst[i]]);
    float ex = __expf(elog[i] - m);
    eexp[i]  = ex;
    atomicAdd(&nden[dst[i]], ex);
}

// M[dst[e],:] += T[src[e],:] * (eexp[e] / (nden[dst[e]] + 1e-16))   (warp = 1 edge)
__global__ void k_scatter_msg(const float* __restrict__ T, const int* __restrict__ srcIdx,
                              const int* __restrict__ dstIdx, const float* __restrict__ eexp,
                              const float* __restrict__ nden, float* __restrict__ Mo, int nE) {
    int g = blockIdx.x * blockDim.x + threadIdx.x;
    int e = g >> 5, c = (g & 31) * 4;
    if (e >= nE) return;
    int s = srcIdx ? srcIdx[e] : e;
    int d = dstIdx[e];
    float alpha = eexp[e] / (nden[d] + 1e-16f);
    const float4 v = *(const float4*)&T[(size_t)s * 128 + c];
    float* o = &Mo[(size_t)d * 128 + c];
    atomicAdd(o + 0, v.x * alpha); atomicAdd(o + 1, v.y * alpha);
    atomicAdd(o + 2, v.z * alpha); atomicAdd(o + 3, v.w * alpha);
}

// plain segment sum (readout init): M[dst[n],:] += T[n,:]
__global__ void k_seg_sum(const float* __restrict__ T, const int* __restrict__ dstIdx,
                          float* __restrict__ Mo, int n) {
    int g = blockIdx.x * blockDim.x + threadIdx.x;
    int e = g >> 5, c = (g & 31) * 4;
    if (e >= n) return;
    int d = dstIdx[e];
    const float4 v = *(const float4*)&T[(size_t)e * 128 + c];
    float* o = &Mo[(size_t)d * 128 + c];
    atomicAdd(o + 0, v.x); atomicAdd(o + 1, v.y);
    atomicAdd(o + 2, v.z); atomicAdd(o + 3, v.w);
}

// GRUCell combine: out = relu((1-z)*n + z*hid)
__global__ void k_gru_combine(const float* __restrict__ gi, const float* __restrict__ gh,
                              const float* __restrict__ hid, float* __restrict__ out, int rows) {
    int idx = blockIdx.x * blockDim.x + threadIdx.x;
    if (idx >= rows * 128) return;
    int r = idx >> 7, j = idx & 127;
    size_t b = (size_t)r * 384;
    float ir = gi[b + j], iz = gi[b + 128 + j], in_ = gi[b + 256 + j];
    float hr = gh[b + j], hz = gh[b + 128 + j], hn  = gh[b + 256 + j];
    float rg = 1.f / (1.f + __expf(-(ir + hr)));
    float zg = 1.f / (1.f + __expf(-(iz + hz)));
    float ng = tanhf(in_ + rg * hn);
    float h  = (1.f - zg) * ng + zg * hid[idx];
    out[idx] = h > 0.f ? h : 0.f;
}

// final: out[m] = sum_c relu(x[m,c]) * w[c] + b   (warp per molecule, K=256)
__global__ void k_mlp2(const float* __restrict__ X, const float* __restrict__ w,
                       const float* __restrict__ b, float* __restrict__ out, int rows) {
    int g = blockIdx.x * blockDim.x + threadIdx.x;
    int row = g >> 5, lane = g & 31;
    if (row >= rows) return;
    float s = 0.f;
    for (int c = lane; c < 256; c += 32) {
        float x = X[(size_t)row * 256 + c];
        x = x > 0.f ? x : 0.f;
        s += x * w[c];
    }
#pragma unroll
    for (int off = 16; off >= 1; off >>= 1) s += __shfl_xor(s, off, 32);
    if (lane == 0) out[row] = s + b[0];
}

// ---------------------------------------------------------------------------
static inline int cdiv(int a, int b) { return (a + b - 1) / b; }

extern "C" void kernel_launch(void* const* d_in, const int* in_sizes, int n_in,
                              void* d_out, int out_size, void* d_ws, size_t ws_size,
                              hipStream_t stream) {
    (void)in_sizes; (void)n_in; (void)out_size; (void)ws_size;
    constexpr int NN = 100000, NE = 400000, NM = 5000;

    const float* x        = (const float*)d_in[0];
    const float* ea       = (const float*)d_in[1];
    const int*   eidx     = (const int*)d_in[2];
    const int*   src      = eidx;
    const int*   dst      = eidx + NE;
    const int*   batch    = (const int*)d_in[3];
    const float* lin1_w   = (const float*)d_in[5];
    const float* lin1_b   = (const float*)d_in[6];
    const float* g_lin1_w = (const float*)d_in[7];
    const float* g_att_l  = (const float*)d_in[8];
    const float* g_att_r  = (const float*)d_in[9];
    const float* g_lin2_w = (const float*)d_in[10];
    const float* g_bias   = (const float*)d_in[11];
    const float* g0_wih   = (const float*)d_in[12];
    const float* g0_whh   = (const float*)d_in[13];
    const float* g0_bih   = (const float*)d_in[14];
    const float* g0_bhh   = (const float*)d_in[15];
    const float* a_lin_w  = (const float*)d_in[16];
    const float* a_asrc   = (const float*)d_in[17];
    const float* a_adst   = (const float*)d_in[18];
    const float* a_bias   = (const float*)d_in[19];
    const float* a_wih    = (const float*)d_in[20];
    const float* a_whh    = (const float*)d_in[21];
    const float* a_bih    = (const float*)d_in[22];
    const float* a_bhh    = (const float*)d_in[23];
    const float* m_lin_w  = (const float*)d_in[24];
    const float* m_asrc   = (const float*)d_in[25];
    const float* m_adst   = (const float*)d_in[26];
    const float* m_bias   = (const float*)d_in[27];
    const float* m_wih    = (const float*)d_in[28];
    const float* m_whh    = (const float*)d_in[29];
    const float* m_bih    = (const float*)d_in[30];
    const float* m_bhh    = (const float*)d_in[31];
    const float* lin2_w   = (const float*)d_in[32];
    const float* lin2_b   = (const float*)d_in[33];
    const float* mlp1_w   = (const float*)d_in[34];
    const float* mlp1_b   = (const float*)d_in[35];
    const float* mlp2_w   = (const float*)d_in[36];
    const float* mlp2_b   = (const float*)d_in[37];

    char* wsp = (char*)d_ws;
    auto alloc = [&](size_t nfloats) -> float* {
        float* p = (float*)wsp;
        wsp += ((nfloats * 4 + 255) / 256) * 256;
        return p;
    };
    float* f_h0   = alloc((size_t)NN * 128);
    float* f_xcur = alloc((size_t)NN * 128);
    float* f_h    = alloc((size_t)NN * 128);   // also the message accumulator m
    float* f_t    = alloc((size_t)NN * 128);   // t / hs / hs_nodes
    float* f_gi   = alloc((size_t)NN * 384);
    float* f_gh   = alloc((size_t)NN * 384);
    float* f_elog = alloc(NE);
    float* f_eexp = alloc(NE);
    unsigned* u_nmax = (unsigned*)alloc(NN);
    float* f_nden = alloc(NN);
    float* f_nd1  = alloc(NN);
    float* f_nd2  = alloc(NN);
    float* f_mout = alloc((size_t)NM * 128);
    float* f_mhd  = alloc((size_t)NM * 128);
    float* f_mh   = alloc((size_t)NM * 128);
    float* f_mgi  = alloc((size_t)NM * 384);
    float* f_mgh  = alloc((size_t)NM * 384);
    float* f_madst = alloc(NM);
    unsigned* u_mmax = (unsigned*)alloc(NM);
    float* f_mden = alloc(NM);
    float* f_l2o  = alloc((size_t)NM * 512);
    float* f_m1o  = alloc((size_t)NM * 256);

    const unsigned NEG_INF_ENC = 0x007FFFFFu;  // fenc(-inf)
    auto E1 = [](int n) { return dim3(cdiv(n, 256)); };
    dim3 B(256);

    // h0 = leaky(x @ lin1^T + b)
    k_gemm<1, false><<<dim3(cdiv(NN, 128)), B, 0, stream>>>(x, lin1_w, lin1_b, f_h0, NN, 128, 44);
    // hdotr = h0 . gate_att_r
    k_node_dot2<<<E1(NN * 32), B, 0, stream>>>(f_h0, g_att_r, nullptr, f_nd1, nullptr, NN);
    // fused GATEConv edge logits
    k_edge_gate_logits<<<dim3(cdiv(NE, 128)), B, 0, stream>>>(f_h0, ea, src, dst, g_lin1_w,
                                                              g_att_l, f_nd1, f_elog, NE);
    // segment softmax over dst
    k_fill_u32<<<E1(NN), B, 0, stream>>>(u_nmax, NEG_INF_ENC, NN);
    k_fill_f32<<<E1(NN), B, 0, stream>>>(f_nden, 0.f, NN);
    k_seg_max<<<E1(NE), B, 0, stream>>>(f_elog, dst, u_nmax, NE);
    k_seg_expsum<<<E1(NE), B, 0, stream>>>(f_elog, dst, u_nmax, f_eexp, f_nden, NE);
    // t = h0 @ gate_lin2^T ; m = bias + scatter(t[src]*alpha) ; h = elu(m)
    k_gemm<0, false><<<dim3(cdiv(NN, 128)), B, 0, stream>>>(f_h0, g_lin2_w, nullptr, f_t, NN, 128, 128);
    k_bias_init<<<E1(NN * 128), B, 0, stream>>>(f_h, g_bias, NN);
    k_scatter_msg<<<E1(NE * 32), B, 0, stream>>>(f_t, src, dst, f_eexp, f_nden, f_h, NE);
    k_elu<<<E1(NN * 128), B, 0, stream>>>(f_h, NN * 128);
    // xcur = relu(gru(h, h0))
    k_gemm<0, false><<<dim3(cdiv(NN, 128)), B, 0, stream>>>(f_h, g0_wih, g0_bih, f_gi, NN, 384, 128);
    k_gemm<0, false><<<dim3(cdiv(NN, 128)), B, 0, stream>>>(f_h0, g0_whh, g0_bhh, f_gh, NN, 384, 128);
    k_gru_combine<<<E1(NN * 128), B, 0, stream>>>(f_gi, f_gh, f_h0, f_xcur, NN);

    // two extra atom GATConv + GRU layers
    for (int l = 0; l < 2; ++l) {
        const float* lw  = a_lin_w + (size_t)l * 128 * 128;
        const float* as_ = a_asrc + (size_t)l * 128;
        const float* ad_ = a_adst + (size_t)l * 128;
        const float* ab_ = a_bias + (size_t)l * 128;
        const float* wih = a_wih + (size_t)l * 384 * 128;
        const float* whh = a_whh + (size_t)l * 384 * 128;
        const float* bih = a_bih + (size_t)l * 384;
        const float* bhh = a_bhh + (size_t)l * 384;
        k_gemm<0, false><<<dim3(cdiv(NN, 128)), B, 0, stream>>>(f_xcur, lw, nullptr, f_t, NN, 128, 128);
        k_node_dot2<<<E1(NN * 32), B, 0, stream>>>(f_t, as_, ad_, f_nd1, f_nd2, NN);
        k_att_logits<<<E1(NE), B, 0, stream>>>(f_nd1, f_nd2, src, dst, f_elog, NE);
        k_fill_u32<<<E1(NN), B, 0, stream>>>(u_nmax, NEG_INF_ENC, NN);
        k_fill_f32<<<E1(NN), B, 0, stream>>>(f_nden, 0.f, NN);
        k_seg_max<<<E1(NE), B, 0, stream>>>(f_elog, dst, u_nmax, NE);
        k_seg_expsum<<<E1(NE), B, 0, stream>>>(f_elog, dst, u_nmax, f_eexp, f_nden, NE);
        k_bias_init<<<E1(NN * 128), B, 0, stream>>>(f_h, ab_, NN);
        k_scatter_msg<<<E1(NE * 32), B, 0, stream>>>(f_t, src, dst, f_eexp, f_nden, f_h, NE);
        k_elu<<<E1(NN * 128), B, 0, stream>>>(f_h, NN * 128);
        k_gemm<0, false><<<dim3(cdiv(NN, 128)), B, 0, stream>>>(f_h, wih, bih, f_gi, NN, 384, 128);
        k_gemm<0, false><<<dim3(cdiv(NN, 128)), B, 0, stream>>>(f_xcur, whh, bhh, f_gh, NN, 384, 128);
        k_gru_combine<<<E1(NN * 128), B, 0, stream>>>(f_gi, f_gh, f_xcur, f_xcur, NN);
    }

    // molecule readout init: out = relu(segsum(xcur, batch))
    k_fill_f32<<<E1(NM * 128), B, 0, stream>>>(f_mout, 0.f, NM * 128);
    k_seg_sum<<<E1(NN * 32), B, 0, stream>>>(f_xcur, batch, f_mout, NN);
    k_relu<<<E1(NM * 128), B, 0, stream>>>(f_mout, NM * 128);
    // hs_nodes (constant across timesteps) and its att_src dot
    k_gemm<0, false><<<dim3(cdiv(NN, 128)), B, 0, stream>>>(f_xcur, m_lin_w, nullptr, f_t, NN, 128, 128);
    k_node_dot2<<<E1(NN * 32), B, 0, stream>>>(f_t, m_asrc, nullptr, f_nd1, nullptr, NN);

    for (int t = 0; t < 3; ++t) {
        k_gemm<0, false><<<dim3(cdiv(NM, 128)), B, 0, stream>>>(f_mout, m_lin_w, nullptr, f_mhd, NM, 128, 128);
        k_node_dot2<<<E1(NM * 32), B, 0, stream>>>(f_mhd, m_adst, nullptr, f_madst, nullptr, NM);
        k_att_logits<<<E1(NN), B, 0, stream>>>(f_nd1, f_madst, nullptr, batch, f_elog, NN);
        k_fill_u32<<<E1(NM), B, 0, stream>>>(u_mmax, NEG_INF_ENC, NM);
        k_fill_f32<<<E1(NM), B, 0, stream>>>(f_mden, 0.f, NM);
        k_seg_max<<<E1(NN), B, 0, stream>>>(f_elog, batch, u_mmax, NN);
        k_seg_expsum<<<E1(NN), B, 0, stream>>>(f_elog, batch, u_mmax, f_eexp, f_mden, NN);
        k_bias_init<<<E1(NM * 128), B, 0, stream>>>(f_mh, m_bias, NM);
        k_scatter_msg<<<E1(NN * 32), B, 0, stream>>>(f_t, nullptr, batch, f_eexp, f_mden, f_mh, NN);
        k_elu<<<E1(NM * 128), B, 0, stream>>>(f_mh, NM * 128);
        k_gemm<0, false><<<dim3(cdiv(NM, 128)), B, 0, stream>>>(f_mh, m_wih, m_bih, f_mgi, NM, 384, 128);
        k_gemm<0, false><<<dim3(cdiv(NM, 128)), B, 0, stream>>>(f_mout, m_whh, m_bhh, f_mgh, NM, 384, 128);
        k_gru_combine<<<E1(NM * 128), B, 0, stream>>>(f_mgi, f_mgh, f_mout, f_mout, NM);
    }

    // head: lin2 -> relu -> mlp1 -> relu -> mlp2
    k_gemm<0, false><<<dim3(cdiv(NM, 128)), B, 0, stream>>>(f_mout, lin2_w, lin2_b, f_l2o, NM, 512, 128);
    k_gemm<0, true><<<dim3(cdiv(NM, 128)), B, 0, stream>>>(f_l2o, mlp1_w, mlp1_b, f_m1o, NM, 256, 512);
    k_mlp2<<<E1(NM * 32), B, 0, stream>>>(f_m1o, mlp2_w, mlp2_b, (float*)d_out, NM);
}
